// MultiHeadedAttention_1297080123446
// MI455X (gfx1250) — matrix operand order repaired and checked by reading, not compiled, not run on hardware
//
#include <hip/hip_runtime.h>
#include <hip/hip_bf16.h>

// MI455X (gfx1250) multi-head attention forward.
// Compute-bound (~450 FLOP/byte): all matmuls via v_wmma_f32_16x16x32_f16,
// flash-attention so S x S scores never hit HBM. f16 storage, f32 accumulation.
// GEMMs use 256x64 block tiles / 32x64 wave tiles: 8 WMMA per 6 fragment loads.

#define B_ 4
#define S_ 2048
#define D_ 1024
#define H_ 16
#define E_ 64

typedef __attribute__((ext_vector_type(16))) _Float16 v16h;
typedef __attribute__((ext_vector_type(8)))  _Float16 v8h;
typedef __attribute__((ext_vector_type(4)))  _Float16 v4h;
typedef __attribute__((ext_vector_type(2)))  _Float16 v2h;
typedef __attribute__((ext_vector_type(8)))  float    v8f;

// ---------------------------------------------------------------------------
// WMMA fragment helpers (layouts per CDNA5 ISA 7.12.2, wave32)
// A (16x32 f16): lane m=l&15, hi=l>>4; elem j<8 -> k=hi*8+j ; j>=8 -> k=16+hi*8+(j-8)
// B (32x16 f16) from Bt[N][K] in LDS: lane n=l&15; elem j<8 -> k=hi*8+j ; j>=8 -> k=16+hi*8+(j-8)
// C/D (16x16 f32): lane n=l&15; vgpr i -> row m=i+8*hi
// ---------------------------------------------------------------------------
__device__ __forceinline__ v16h load_a_frag(const _Float16* base, int lda) {
  const int lane = threadIdx.x & 31;
  const int m = lane & 15, hi = lane >> 4;
  const _Float16* r = base + (size_t)m * lda + hi * 8;
  v8h lo = *(const v8h*)(r);
  v8h hh = *(const v8h*)(r + 16);
  v16h a;
#pragma unroll
  for (int j = 0; j < 8; ++j) { a[j] = lo[j]; a[j + 8] = hh[j]; }
  return a;
}

__device__ __forceinline__ v16h load_b_frag(const _Float16* bt, int ldb) {
  const int lane = threadIdx.x & 31;
  const int n = lane & 15, hi = lane >> 4;
  const _Float16* r = bt + (size_t)n * ldb + hi * 8;
  v8h lo = *(const v8h*)(r);
  v8h hh = *(const v8h*)(r + 16);
  v16h b;
#pragma unroll
  for (int j = 0; j < 8; ++j) { b[j] = lo[j]; b[j + 8] = hh[j]; }
  return b;
}

__device__ __forceinline__ v8f wmma16(v16h a, v16h b, v8f c) {
  return __builtin_amdgcn_wmma_f32_16x16x32_f16(false, a, false, b, (short)0, c,
                                                false, false);
}

// ---------------------------------------------------------------------------
// fp32 -> fp16 conversion (vectorized x4)
// ---------------------------------------------------------------------------
__global__ void cvt_f32_f16(const float* __restrict__ src,
                            _Float16* __restrict__ dst, int n4) {
  int i = blockIdx.x * blockDim.x + threadIdx.x;
  if (i < n4) {
    float4 v = ((const float4*)src)[i];
    v4h o;
    o[0] = (_Float16)v.x; o[1] = (_Float16)v.y;
    o[2] = (_Float16)v.z; o[3] = (_Float16)v.w;
    ((v4h*)dst)[i] = o;
  }
}

// ---------------------------------------------------------------------------
// QKV projection: Out[b,h,s,e] = sum_d X[b,s,d] * W[h,d,e] + bias[h,e]
// Block tile 256x64, 8 waves each 32x64, K-step 32 -> 8 WMMA / wave / step.
// ---------------------------------------------------------------------------
__global__ __launch_bounds__(256) void qkv_proj(
    const _Float16* __restrict__ X,    // [B,S,D] f16
    const _Float16* __restrict__ W,    // [H,D,E] f16
    const float* __restrict__ bias,    // [H,E]
    _Float16* __restrict__ Out) {      // [B,H,S,E] f16
  __shared__ __align__(16) _Float16 As[256 * 32];
  __shared__ __align__(16) _Float16 Bt[64 * 32];
  const int t = threadIdx.x;
  const int b = blockIdx.z, h = blockIdx.y;
  const int row0 = blockIdx.x * 256;
  const int w = t >> 5, lane = t & 31;
  v8f acc[2][4] = {};
  const size_t xbase = ((size_t)b * S_ + row0) * D_;
  const size_t wbase = (size_t)h * D_ * E_;

  for (int k0 = 0; k0 < D_; k0 += 32) {
    {  // A tile: 256 rows x 32 halves (two half-rows per thread)
      int rr = t >> 1, c = (t & 1) * 16;
#pragma unroll
      for (int it = 0; it < 2; ++it) {
        int r = rr + it * 128;
        const _Float16* src = X + xbase + (size_t)r * D_ + k0 + c;
        _Float16* dst = As + r * 32 + c;
        *(float4*)dst = *(const float4*)src;
        *(float4*)(dst + 8) = *(const float4*)(src + 8);
      }
    }
    {  // B tile transposed, packed pair-writes: Bt[n][kk] = W[h, k0+kk, n]
      int kk2 = (t & 15) * 2, n0 = (t >> 4) * 4;
      v4h wa = *(const v4h*)(W + wbase + (size_t)(k0 + kk2) * E_ + n0);
      v4h wb = *(const v4h*)(W + wbase + (size_t)(k0 + kk2 + 1) * E_ + n0);
#pragma unroll
      for (int i = 0; i < 4; ++i) {
        v2h pr; pr[0] = wa[i]; pr[1] = wb[i];
        *(v2h*)(Bt + (n0 + i) * 32 + kk2) = pr;
      }
    }
    __syncthreads();
    v16h a0 = load_a_frag(As + (w * 32) * 32, 32);
    v16h a1 = load_a_frag(As + (w * 32 + 16) * 32, 32);
#pragma unroll
    for (int nt = 0; nt < 4; ++nt) {
      v16h bf = load_b_frag(Bt + nt * 16 * 32, 32);
      acc[0][nt] = wmma16(a0, bf, acc[0][nt]);
      acc[1][nt] = wmma16(a1, bf, acc[1][nt]);
    }
    __syncthreads();
  }

  const int n = lane & 15, hi = lane >> 4;
#pragma unroll
  for (int mt = 0; mt < 2; ++mt)
#pragma unroll
    for (int nt = 0; nt < 4; ++nt) {
      int col = nt * 16 + n;
      float bi = bias[h * E_ + col];
#pragma unroll
      for (int i = 0; i < 8; ++i) {
        int row = row0 + w * 32 + mt * 16 + i + 8 * hi;
        Out[((size_t)(b * H_ + h) * S_ + row) * E_ + col] =
            (_Float16)(acc[mt][nt][i] + bi);
      }
    }
}

// ---------------------------------------------------------------------------
// Flash attention: per wave one 16-row query strip, key/value tiles of 32,
// online softmax, O accumulated in f32, written as f16 [B,S,H*E] (head-concat).
// ---------------------------------------------------------------------------
__global__ __launch_bounds__(256) void flash_attn(
    const _Float16* __restrict__ Q,    // [B,H,S,E]
    const _Float16* __restrict__ Kp,   // [B,H,S,E]
    const _Float16* __restrict__ Vp,   // [B,H,S,E]
    _Float16* __restrict__ O) {        // [B,S,H*E]
  __shared__ __align__(16) _Float16 Kt[32 * 64];     // Kt[key][e]  (Bt for Q*K^T)
  __shared__ __align__(16) _Float16 Vt[64 * 32];     // Vt[e][key]  (Bt for P*V)
  __shared__ __align__(16) _Float16 Ps[8][16 * 32];  // per-wave P relayout
  const int t = threadIdx.x, w = t >> 5, lane = t & 31;
  const int b = blockIdx.z, h = blockIdx.y;
  const int qrow0 = blockIdx.x * 128 + w * 16;
  const int n = lane & 15, hi = lane >> 4;
  const size_t bh = (size_t)(b * H_ + h) * S_;

  // Q fragments in registers for the whole kernel; fold in 1/sqrt(E)=0.125
  // (exact power of two in f16, so no precision loss).
  const _Float16* qbase = Q + (bh + qrow0) * E_;
  v16h aQ0 = load_a_frag(qbase, E_);
  v16h aQ1 = load_a_frag(qbase + 32, E_);
#pragma unroll
  for (int j = 0; j < 16; ++j) {
    aQ0[j] = aQ0[j] * (_Float16)0.125f;
    aQ1[j] = aQ1[j] * (_Float16)0.125f;
  }

  v8f oacc[4] = {};
  float mrow[8], lrow[8];
#pragma unroll
  for (int i = 0; i < 8; ++i) { mrow[i] = -1e30f; lrow[i] = 0.f; }

  for (int kt = 0; kt < S_; kt += 32) {
    {  // K tile: rows are already the Bt layout for scores
      int key = t >> 3, c = (t & 7) * 8;
      *(float4*)(Kt + key * 64 + c) =
          *(const float4*)(Kp + (bh + kt + key) * E_ + c);
    }
    {  // V tile transposed, packed pair-writes: Vt[e][key]
      int key2 = (t & 15) * 2, e0 = (t >> 4) * 4;
      v4h va = *(const v4h*)(Vp + (bh + kt + key2) * E_ + e0);
      v4h vb = *(const v4h*)(Vp + (bh + kt + key2 + 1) * E_ + e0);
#pragma unroll
      for (int i = 0; i < 4; ++i) {
        v2h pr; pr[0] = va[i]; pr[1] = vb[i];
        *(v2h*)(Vt + (e0 + i) * 32 + key2) = pr;
      }
    }
    __syncthreads();

    // scores: 16 rows x 32 keys (scale already folded into Q)
    v8f sc[2];
#pragma unroll
    for (int ntile = 0; ntile < 2; ++ntile) {
      v8f c = {};
      c = wmma16(aQ0, load_b_frag(Kt + ntile * 16 * 64, 64), c);
      c = wmma16(aQ1, load_b_frag(Kt + ntile * 16 * 64 + 32, 64), c);
      sc[ntile] = c;
    }

    // online softmax: row stats live in the 16-lane group that owns each row
    float p0[8], p1[8], alpha[8];
#pragma unroll
    for (int i = 0; i < 8; ++i) {
      float v = fmaxf(sc[0][i], sc[1][i]);
#pragma unroll
      for (int mask = 1; mask < 16; mask <<= 1)
        v = fmaxf(v, __shfl_xor(v, mask, 32));
      float nm = fmaxf(mrow[i], v);
      alpha[i] = __expf(mrow[i] - nm);
      mrow[i] = nm;
      p0[i] = __expf(sc[0][i] - nm);
      p1[i] = __expf(sc[1][i] - nm);
      float rs = p0[i] + p1[i];
#pragma unroll
      for (int mask = 1; mask < 16; mask <<= 1)
        rs += __shfl_xor(rs, mask, 32);
      lrow[i] = lrow[i] * alpha[i] + rs;
    }
#pragma unroll
    for (int ntile = 0; ntile < 4; ++ntile)
#pragma unroll
      for (int i = 0; i < 8; ++i) oacc[ntile][i] *= alpha[i];

    // re-layout P (C-form -> A-form) through per-wave LDS scratch
    _Float16* pw = &Ps[w][0];
#pragma unroll
    for (int i = 0; i < 8; ++i) {
      pw[(i + 8 * hi) * 32 + n] = (_Float16)p0[i];
      pw[(i + 8 * hi) * 32 + 16 + n] = (_Float16)p1[i];
    }
    v16h aP = load_a_frag(pw, 32);
#pragma unroll
    for (int ntile = 0; ntile < 4; ++ntile)
      oacc[ntile] = wmma16(aP, load_b_frag(Vt + ntile * 16 * 32, 32), oacc[ntile]);
    __syncthreads();
  }

  // normalize and store head-concatenated f16 output
#pragma unroll
  for (int i = 0; i < 8; ++i) lrow[i] = 1.f / lrow[i];
#pragma unroll
  for (int ntile = 0; ntile < 4; ++ntile) {
    int d = h * E_ + ntile * 16 + n;
#pragma unroll
    for (int i = 0; i < 8; ++i) {
      int row = qrow0 + i + 8 * hi;
      O[((size_t)b * S_ + row) * D_ + d] = (_Float16)(oacc[ntile][i] * lrow[i]);
    }
  }
}

// ---------------------------------------------------------------------------
// Output projection: out[m,n] = sum_k O16[m,k] * Wo[n,k] + bo[n]  (f32 out).
// Wo rows are already the Bt[N][K] layout -> no transpose on LDS store.
// Block tile 256x64, wave tile 32x64.
// ---------------------------------------------------------------------------
__global__ __launch_bounds__(256) void out_proj(
    const _Float16* __restrict__ A,    // [B*S, D] f16
    const _Float16* __restrict__ Wt,   // [D, D] f16 (Wo row-major [n][k])
    const float* __restrict__ bo,      // [D]
    float* __restrict__ out) {         // [B*S, D] f32
  __shared__ __align__(16) _Float16 As[256 * 32];
  __shared__ __align__(16) _Float16 Bt[64 * 32];
  const int t = threadIdx.x, w = t >> 5, lane = t & 31;
  const int m0 = blockIdx.x * 256;
  const int n0g = blockIdx.y * 64;
  v8f acc[2][4] = {};

  for (int k0 = 0; k0 < D_; k0 += 32) {
    {  // A tile
      int rr = t >> 1, c = (t & 1) * 16;
#pragma unroll
      for (int it = 0; it < 2; ++it) {
        int r = rr + it * 128;
        const _Float16* src = A + (size_t)(m0 + r) * D_ + k0 + c;
        _Float16* dst = As + r * 32 + c;
        *(float4*)dst = *(const float4*)src;
        *(float4*)(dst + 8) = *(const float4*)(src + 8);
      }
    }
    {  // B tile: direct rows of Wo
      int nn = t >> 2, c = (t & 3) * 8;
      *(float4*)(Bt + nn * 32 + c) =
          *(const float4*)(Wt + (size_t)(n0g + nn) * D_ + k0 + c);
    }
    __syncthreads();
    v16h a0 = load_a_frag(As + (w * 32) * 32, 32);
    v16h a1 = load_a_frag(As + (w * 32 + 16) * 32, 32);
#pragma unroll
    for (int nt = 0; nt < 4; ++nt) {
      v16h bf = load_b_frag(Bt + nt * 16 * 32, 32);
      acc[0][nt] = wmma16(a0, bf, acc[0][nt]);
      acc[1][nt] = wmma16(a1, bf, acc[1][nt]);
    }
    __syncthreads();
  }

  const int n = lane & 15, hi = lane >> 4;
#pragma unroll
  for (int mt = 0; mt < 2; ++mt)
#pragma unroll
    for (int nt = 0; nt < 4; ++nt) {
      int col = n0g + nt * 16 + n;
      float bi = bo[col];
#pragma unroll
      for (int i = 0; i < 8; ++i) {
        int row = m0 + w * 32 + mt * 16 + i + 8 * hi;
        out[(size_t)row * D_ + col] = acc[mt][nt][i] + bi;
      }
    }
}

// ---------------------------------------------------------------------------
extern "C" void kernel_launch(void* const* d_in, const int* in_sizes, int n_in,
                              void* d_out, int out_size, void* d_ws,
                              size_t ws_size, hipStream_t stream) {
  const float* q  = (const float*)d_in[0];
  const float* k  = (const float*)d_in[1];
  const float* v  = (const float*)d_in[2];
  const float* Wq = (const float*)d_in[3];
  const float* bq = (const float*)d_in[4];
  const float* Wk = (const float*)d_in[5];
  const float* bk = (const float*)d_in[6];
  const float* Wv = (const float*)d_in[7];
  const float* bv = (const float*)d_in[8];
  const float* Wo = (const float*)d_in[9];
  const float* bo = (const float*)d_in[10];
  float* out = (float*)d_out;

  const size_t NX = (size_t)B_ * S_ * D_;  // 8,388,608 activation elems
  const size_t NW = (size_t)H_ * D_ * E_;  // 1,048,576 per-head weight elems

  char* ws = (char*)d_ws;
  size_t off = 0;
  auto alloc = [&](size_t bytes) {
    char* p = ws + off;
    off += (bytes + 255) & ~(size_t)255;
    return p;
  };
  _Float16* q16  = (_Float16*)alloc(NX * 2);
  _Float16* k16  = (_Float16*)alloc(NX * 2);
  _Float16* v16  = (_Float16*)alloc(NX * 2);
  _Float16* Wq16 = (_Float16*)alloc(NW * 2);
  _Float16* Wk16 = (_Float16*)alloc(NW * 2);
  _Float16* Wv16 = (_Float16*)alloc(NW * 2);
  _Float16* Wo16 = (_Float16*)alloc((size_t)D_ * D_ * 2);
  _Float16* Qp   = (_Float16*)alloc(NX * 2);
  _Float16* Kp   = (_Float16*)alloc(NX * 2);
  _Float16* Vp   = (_Float16*)alloc(NX * 2);
  _Float16* O16  = (_Float16*)alloc(NX * 2);
  (void)ws_size;

  auto cvt = [&](const float* s, _Float16* d, size_t nelem) {
    int n4 = (int)(nelem / 4);
    cvt_f32_f16<<<(n4 + 255) / 256, 256, 0, stream>>>(s, d, n4);
  };
  cvt(q, q16, NX);
  cvt(k, k16, NX);
  cvt(v, v16, NX);
  cvt(Wq, Wq16, NW);
  cvt(Wk, Wk16, NW);
  cvt(Wv, Wv16, NW);
  cvt(Wo, Wo16, (size_t)D_ * D_);

  dim3 gProj(S_ / 256, H_, B_);
  qkv_proj<<<gProj, 256, 0, stream>>>(q16, Wq16, bq, Qp);
  qkv_proj<<<gProj, 256, 0, stream>>>(k16, Wk16, bk, Kp);
  qkv_proj<<<gProj, 256, 0, stream>>>(v16, Wv16, bv, Vp);

  dim3 gAttn(S_ / 128, H_, B_);
  flash_attn<<<gAttn, 256, 0, stream>>>(Qp, Kp, Vp, O16);

  dim3 gOut((B_ * S_) / 256, D_ / 64);
  out_proj<<<gOut, 256, 0, stream>>>(O16, Wo16, bo, out);
}
